// LSTM_48730698940425
// MI455X (gfx1250) — matrix operand order repaired and checked by reading, not compiled, not run on hardware
//
#include <hip/hip_runtime.h>
#include <hip/hip_bf16.h>

typedef __bf16 bf16;
typedef __attribute__((ext_vector_type(16))) __bf16 v16bf;
typedef __attribute__((ext_vector_type(8)))  __bf16 v8bf;
typedef __attribute__((ext_vector_type(8)))  float   v8f;
typedef __attribute__((ext_vector_type(4)))  unsigned int v4u;
typedef __attribute__((ext_vector_type(8)))  int v8i;
typedef __attribute__((ext_vector_type(4)))  int v4i;

#define S_LEN 2048
#define BATCH 16
#define IDIM  512
#define HDIM  512
#define NGATE 3
#define NWG2  16        // phase-2 workgroups (one H-slice of 32 cols each)
#define HROW  520       // padded LDS row stride in bf16 (1040 B: +16 B pad per 1024 B row)

// workspace layout (bytes); total ~144 MB
#define WS_BAR   0ull
#define WS_HBUF  4096ull                    // 2 * 16*512 bf16 (double-buffered h)
#define WS_WT    65536ull                   // 1536*512 bf16 = 1.5 MB (W transposed, n-major)
#define WS_RT    (WS_WT + (2ull<<20))       // 1.5 MB (R transposed, n-major)
#define WS_XBF   (8ull<<20)                 // x in bf16, 32 MB
#define WS_XG    (48ull<<20)                // gate preactivations bf16 (3,S,B,H), 96 MB

// ---------------- init: zero barrier + h double-buffer ----------------
__global__ void lstm_init(unsigned* bar, bf16* hbuf) {
    int t = blockIdx.x * blockDim.x + threadIdx.x;
    if (t < 2) bar[t] = 0u;
    for (int i = t; i < 2 * BATCH * HDIM; i += blockDim.x * gridDim.x)
        hbuf[i] = (bf16)0.0f;
}

// ---------------- fp32 -> bf16 for x ----------------
__global__ void cvt_x(const float* __restrict__ x, bf16* __restrict__ xbf, int n) {
    int i = blockIdx.x * blockDim.x + threadIdx.x;
    if (i < n) xbf[i] = (bf16)x[i];
}

// ---------------- build transposed bf16 weights: WT[g*512+n][k] = W_g[k][n] ----------------
__global__ void cvt_w(const float* __restrict__ Wi, const float* __restrict__ Wo,
                      const float* __restrict__ Wz, const float* __restrict__ Ri,
                      const float* __restrict__ Ro, const float* __restrict__ Rz,
                      bf16* __restrict__ WT, bf16* __restrict__ RT) {
    int i = blockIdx.x * blockDim.x + threadIdx.x;   // over 3*512*512
    if (i >= NGATE * HDIM * IDIM) return;
    int k = i & (IDIM - 1);
    int n = (i >> 9) & (HDIM - 1);
    int g = i >> 18;
    const float* W = (g == 0) ? Wi : ((g == 1) ? Wo : Wz);
    const float* R = (g == 0) ? Ri : ((g == 1) ? Ro : Rz);
    WT[i] = (bf16)W[k * HDIM + n];
    RT[i] = (bf16)R[k * HDIM + n];
}

// pack two contiguous 8-elem bf16 chunks into the 16-elem WMMA A operand
__device__ __forceinline__ v16bf pack_a(const bf16* p0, const bf16* p1) {
    v8bf lo = *(const v8bf*)p0;
    v8bf hi = *(const v8bf*)p1;
    v16bf r;
#pragma unroll
    for (int i = 0; i < 8; ++i) { r[i] = lo[i]; r[i + 8] = hi[i]; }
    return r;
}

__device__ __forceinline__ float fast_sigmoid(float x) {
    return 1.0f / (1.0f + __expf(-x));
}
__device__ __forceinline__ float fast_tanh(float x) {
    float e = __expf(2.0f * x);          // saturates correctly at +/-inf
    return 1.0f - 2.0f / (e + 1.0f);
}

// ---------------- phase 1: xg[g,s,b,h] = bf16(x @ W_g + b_g) ----------------
// block = 256 thr (8 waves); wave computes a 16(M) x 64(N) tile, K=512 in 16 WMMA k-steps
__global__ __launch_bounds__(256)
void lstm_xproj(const bf16* __restrict__ xbf, const bf16* __restrict__ WT,
                const float* __restrict__ bi, const float* __restrict__ bo,
                const float* __restrict__ bz, bf16* __restrict__ xg) {
    const int lane  = threadIdx.x & 31;
    const int wave  = threadIdx.x >> 5;
    const int mt    = blockIdx.y * 8 + wave;     // 16-row M tile, 0..2047
    const int nbase = blockIdx.x * 64;           // 0..1535 (gates concatenated)
    const int g     = nbase >> 9;
    const int col0  = nbase & 511;
    const int m15   = lane & 15;
    const int ka    = (lane >> 4) * 16;          // A per-lane-half k offset
    const int kb    = (lane >> 4) * 16;          // B per-lane-half k offset

    const bf16* Arow = xbf + (size_t)(mt * 16 + m15) * IDIM;
    v8f acc[4] = {};
#pragma unroll 4
    for (int kk = 0; kk < IDIM / 32; ++kk) {
        const int k0 = kk * 32;
        v16bf A = pack_a(Arow + k0 + ka, Arow + k0 + ka + 8);
#pragma unroll
        for (int nt = 0; nt < 4; ++nt) {
            const bf16* Bp = WT + (size_t)(nbase + nt * 16 + m15) * IDIM + k0 + kb;
            v16bf Bv = *(const v16bf*)Bp;
            acc[nt] = __builtin_amdgcn_wmma_f32_16x16x32_bf16(
                false, A, false, Bv, (short)0, acc[nt], false, false);
        }
    }
    const float* bias = (g == 0) ? bi : ((g == 1) ? bo : bz);
#pragma unroll
    for (int nt = 0; nt < 4; ++nt) {
        const int col = col0 + nt * 16 + m15;
        const float bv = bias[col];
#pragma unroll
        for (int j = 0; j < 8; ++j) {
            const int m = mt * 16 + j + ((lane >> 4) << 3);  // C/D row layout
            const int b = m >> 11, s = m & 2047;             // m = b*S + s
            xg[(((size_t)g * S_LEN + s) * BATCH + b) * HDIM + col] = (bf16)(acc[nt][j] + bv);
        }
    }
}

// ---------------- device-wide barrier over NWG2 persistent blocks ----------------
__device__ __forceinline__ void grid_sync(unsigned* bar, unsigned target) {
    __threadfence();        // release h writes
    __syncthreads();
    if (threadIdx.x == 0) {
        unsigned old = __hip_atomic_fetch_add(&bar[0], 1u, __ATOMIC_ACQ_REL,
                                              __HIP_MEMORY_SCOPE_AGENT);
        if (old == NWG2 - 1) {
            __hip_atomic_store(&bar[0], 0u, __ATOMIC_RELAXED, __HIP_MEMORY_SCOPE_AGENT);
            __hip_atomic_fetch_add(&bar[1], 1u, __ATOMIC_RELEASE, __HIP_MEMORY_SCOPE_AGENT);
        } else {
            while (__hip_atomic_load(&bar[1], __ATOMIC_ACQUIRE,
                                     __HIP_MEMORY_SCOPE_AGENT) < target)
                __builtin_amdgcn_s_sleep(2);
        }
    }
    __syncthreads();
    __threadfence();        // acquire other WGs' h writes
}

// ---------------- phase 2: persistent scan. 16 WGs x 6 waves. ----------------
// R weights live in VGPRs for the whole scan. Per step, the TDM DMAs the 16KB
// h matrix into LDS once per WG (with pad-on-load so the 16 A rows land 4
// banks apart: conflict-free ds_load_b128), then 6 waves run 16 WMMAs each.
__global__ __launch_bounds__(192)
void lstm_scan(const bf16* __restrict__ RT, const bf16* __restrict__ xg,
               bf16* __restrict__ hbuf, float* __restrict__ out, unsigned* bar) {
    __shared__ bf16  hlds[BATCH * HROW];       // 16 rows x 1040 B (TDM-padded)
    __shared__ float pre[NGATE][BATCH][32];
    __shared__ float cst[BATCH][32];

    const int wg   = blockIdx.x;              // 0..15 -> H cols [wg*32, wg*32+32)
    const int lane = threadIdx.x & 31;
    const int wave = threadIdx.x >> 5;        // 0..5
    const int g    = wave >> 1;
    const int t    = wave & 1;
    const int n15  = lane & 15;
    const int ka   = (lane >> 4) * 16;
    const int kb   = (lane >> 4) * 16;
    const int colg = wg * 32 + t * 16 + n15;  // column within gate (B-operand lane col)

    for (int e = threadIdx.x; e < BATCH * 32; e += blockDim.x)
        (&cst[0][0])[e] = 0.0f;

    // resident recurrent weights: one 32x16 B tile per k-step, 16 k-steps
    v16bf Bw[16];
#pragma unroll
    for (int kk = 0; kk < 16; ++kk)
        Bw[kk] = *(const v16bf*)(RT + (size_t)(g * HDIM + colg) * HDIM + kk * 32 + kb);

    // --- TDM descriptors: 2D tensor 512(elem) x 16(rows), bf16, one per h buffer ---
    const unsigned ldsoff = (unsigned)(size_t)(&hlds[0]);  // flat LDS addr low bits = offset
    // group1: data_size=1(2B) | pad_enable | pad_interval=7 (256 DW) | pad_amount=3 (4 DW)
    v8i g1;
    g1[0] = (1 << 16) | (1 << 20) | (7 << 22) | (3 << 25);
    g1[1] = (int)(512u << 16);      // tensor_dim0[15:0]  -> bits 63:48
    g1[2] = (int)(16u  << 16);      // tensor_dim0 hi = 0; tensor_dim1[15:0]=16 -> bits 95:80
    g1[3] = (int)(512u << 16);      // tensor_dim1 hi = 0; tile_dim0=512 -> bits 127:112
    g1[4] = 16;                     // tile_dim1=16; tile_dim2=0
    g1[5] = 512;                    // tensor_dim0_stride (elements)
    g1[6] = 0;
    g1[7] = 0;
    const v4i gz4 = {0, 0, 0, 0};
    const v8i gz8 = {0, 0, 0, 0, 0, 0, 0, 0};
    v4u g0p[2];
#pragma unroll
    for (int p = 0; p < 2; ++p) {
        unsigned long long ga = (unsigned long long)(size_t)(hbuf + p * (BATCH * HDIM));
        v4u d;
        d[0] = 1u;                                   // count=1, user descriptor
        d[1] = ldsoff;                               // lds_addr
        d[2] = (unsigned)(ga & 0xFFFFFFFFu);         // global_addr[31:0]
        d[3] = (unsigned)((ga >> 32) & 0x01FFFFFFu)  // global_addr[56:32]
             | (2u << 30);                           // type = 2 ("image")
        g0p[p] = d;
    }

    __syncthreads();

    for (int s = 0; s < S_LEN; ++s) {
        v8f acc = {};
        if (s > 0) {  // h_{-1} = 0: skip the GEMM for step 0
            if (wave == 0) {     // TDM issue is per-wave; one wave drives the DMA
                __builtin_amdgcn_tensor_load_to_lds(g0p[(s + 1) & 1], g1, gz4, gz4, gz8, 0);
                __builtin_amdgcn_s_wait_tensorcnt(0);
            }
            __syncthreads();     // h tile now in LDS for all 6 waves
            const bf16* hp = hlds + n15 * HROW;   // padded row: banks shift 4/row
#pragma unroll
            for (int kk = 0; kk < 16; ++kk) {
                v16bf A = pack_a(hp + kk * 32 + ka, hp + kk * 32 + ka + 8);
                acc = __builtin_amdgcn_wmma_f32_16x16x32_bf16(
                    false, A, false, Bw[kk], (short)0, acc, false, false);
            }
        }
        // prefetch next step's input projection rows while this step finishes
        if (s + 1 < S_LEN) {
#pragma unroll
            for (int j = 0; j < 8; ++j) {
                const int b = j + ((lane >> 4) << 3);
                __builtin_prefetch(
                    &xg[(((size_t)g * S_LEN + (s + 1)) * BATCH + b) * HDIM + colg], 0, 0);
            }
        }
        // add input projection, publish gate preactivations to LDS
#pragma unroll
        for (int j = 0; j < 8; ++j) {
            const int b = j + ((lane >> 4) << 3);
            const float xv =
                (float)xg[(((size_t)g * S_LEN + s) * BATCH + b) * HDIM + colg];
            pre[g][b][t * 16 + n15] = acc[j] + xv;
        }
        __syncthreads();
        // elementwise LSTM update on this WG's 32-column slice
        for (int e = threadIdx.x; e < BATCH * 32; e += blockDim.x) {
            const int b = e >> 5, cc = e & 31;
            const float ig = fast_sigmoid(pre[0][b][cc]);
            const float og = fast_sigmoid(pre[1][b][cc]);
            const float zg = fast_tanh(pre[2][b][cc]);
            const float c  = cst[b][cc] + ig * zg;   // no forget gate (per reference)
            cst[b][cc] = c;
            const float h  = og * fast_tanh(c);
            const int hc = wg * 32 + cc;
            out[((size_t)s * BATCH + b) * HDIM + hc] = h;
            hbuf[(s & 1) * (BATCH * HDIM) + b * HDIM + hc] = (bf16)h;
            if (s == S_LEN - 1) {
                out[(size_t)S_LEN * BATCH * HDIM + b * HDIM + hc] = h;                // h_t
                out[(size_t)S_LEN * BATCH * HDIM + BATCH * HDIM + b * HDIM + hc] = c; // c_t
            }
        }
        grid_sync(bar, (unsigned)(s + 1));  // one barrier/step (h double-buffered)
    }
}

extern "C" void kernel_launch(void* const* d_in, const int* in_sizes, int n_in,
                              void* d_out, int out_size, void* d_ws, size_t ws_size,
                              hipStream_t stream) {
    const float* x  = (const float*)d_in[0];
    const float* Wi = (const float*)d_in[1];
    const float* Wo = (const float*)d_in[2];
    const float* Wz = (const float*)d_in[3];
    const float* Ri = (const float*)d_in[4];
    const float* Ro = (const float*)d_in[5];
    const float* Rz = (const float*)d_in[6];
    const float* bi = (const float*)d_in[7];
    const float* bo = (const float*)d_in[8];
    const float* bz = (const float*)d_in[9];

    char* ws = (char*)d_ws;
    unsigned* bar = (unsigned*)(ws + WS_BAR);
    bf16* hbuf = (bf16*)(ws + WS_HBUF);
    bf16* WT   = (bf16*)(ws + WS_WT);
    bf16* RT   = (bf16*)(ws + WS_RT);
    bf16* xbf  = (bf16*)(ws + WS_XBF);
    bf16* xg   = (bf16*)(ws + WS_XG);
    float* out = (float*)d_out;

    lstm_init<<<64, 256, 0, stream>>>(bar, hbuf);

    const int nx = BATCH * S_LEN * IDIM;
    cvt_x<<<(nx + 255) / 256, 256, 0, stream>>>(x, xbf, nx);

    const int nw = NGATE * HDIM * IDIM;
    cvt_w<<<(nw + 255) / 256, 256, 0, stream>>>(Wi, Wo, Wz, Ri, Ro, Rz, WT, RT);

    dim3 g1(24, 256);  // 1536 cols / 64 per block, 32768 rows / 128 per block
    lstm_xproj<<<g1, 256, 0, stream>>>(xbf, WT, bi, bo, bz, xg);

    lstm_scan<<<NWG2, 192, 0, stream>>>(RT, xg, hbuf, out, bar);
}
